// GNN_AttentiveFP_12962211299799
// MI455X (gfx1250) — matrix operand order repaired and checked by reading, not compiled, not run on hardware
//
#include <hip/hip_runtime.h>
#include <hip/hip_bf16.h>
#include <cstdint>

// ---------------------------------------------------------------------------
// AttentiveFP GNN forward for gfx1250 (MI455X).
// Dense GEMMs: v_wmma_f32_16x16x32_f16, fp32 accumulation.
// f16 staging with K-padded (mult-of-32) strides -> unconditional b128 loads.
// Edge-level GEMMs use multi-column-tile waves (A loaded once, 5-6 WMMAs per
// K-step) since their A operands are HBM-resident (mjh = 153 MB).
// N=50000, E=800000, G=1024 are multiples of 16 -> no row guards.
// Workspace ~270 MB.
// ---------------------------------------------------------------------------

typedef __attribute__((ext_vector_type(16))) _Float16 v16h;
typedef __attribute__((ext_vector_type(8)))  _Float16 v8h;
typedef __attribute__((ext_vector_type(8)))  float    v8f;

#define ACT_NONE  0
#define ACT_LRELU 1
#define ACT_RELU  2

#define HPAD 96     // 75 padded to mult of 32
#define GPAD 240    // 225 padded
#define EPAD 32     // 16 padded

__device__ __forceinline__ float lrelu_f(float v) { return v >= 0.f ? v : 0.01f * v; }
__device__ __forceinline__ float sigmoid_f(float v) { return 1.f / (1.f + __expf(-v)); }

__device__ __forceinline__ void atomicMaxFloat(float* addr, float val) {
  if (val >= 0.f) atomicMax((int*)addr, __float_as_int(val));
  else            atomicMin((unsigned int*)addr, (unsigned int)__float_as_int(val));
}

__device__ __forceinline__ v16h loadAfrag(const _Float16* __restrict__ aptr, int kb) {
  v8h c0 = *(const v8h*)(aptr + kb);        // K = base+0..7
  v8h c1 = *(const v8h*)(aptr + kb + 8);    // K = base+8..15
  return __builtin_shufflevector(c0, c1, 0,1,2,3,4,5,6,7,8,9,10,11,12,13,14,15);
}

// ----- single-column-tile accumulation (pointers pre-offset by hi) ----------
__device__ __forceinline__ v8f wmmaAccum(const _Float16* __restrict__ aptr,
                                         const _Float16* __restrict__ wrow,
                                         int Kpad, v8f acc) {
  for (int kb = 0; kb < Kpad; kb += 32) {
    v16h a = loadAfrag(aptr, kb);
    v16h b = *(const v16h*)(wrow + kb);
    acc = __builtin_amdgcn_wmma_f32_16x16x32_f16(false, a, false, b,
                                                 (short)0, acc, false, false);
  }
  return acc;
}

// ----- NT-column-tile accumulation: A fragment loaded once per K-step -------
// wcol = Wh + m*Kpad + (hi?16:0); tile t's column strip at +t*16*Kpad.
template <int NT>
__device__ __forceinline__ void wmmaAccumMulti(const _Float16* __restrict__ aptr,
                                               const _Float16* __restrict__ wcol,
                                               int Kpad, v8f (&acc)[NT]) {
  for (int kb = 0; kb < Kpad; kb += 32) {
    v16h a = loadAfrag(aptr, kb);
#pragma unroll
    for (int t = 0; t < NT; ++t) {
      v16h b = *(const v16h*)(wcol + (size_t)t * 16 * Kpad + kb);
      acc[t] = __builtin_amdgcn_wmma_f32_16x16x32_f16(false, a, false, b,
                                                      (short)0, acc[t], false, false);
    }
  }
}

// ---------------- generic GEMM: f16 A, f16 W, f16 out -----------------------
// grid = (ceil(M/64), ldc/16); block 128.  Weights packed [>=ldc rows, Kpad].
__global__ void __launch_bounds__(128)
gemm_f16_kernel(const _Float16* __restrict__ Ah, int lda, int M, int Kpad,
                const _Float16* __restrict__ Wh, const float* __restrict__ bias,
                int Nout, int act, _Float16* __restrict__ Ch, int ldc) {
  const int lane = threadIdx.x & 31;
  const int m = lane & 15;
  const bool hi = lane >= 16;
  const int rowBase = (blockIdx.x * 4 + (threadIdx.x >> 5)) * 16;
  const int colBase = blockIdx.y * 16;
  if (rowBase >= M) return;
  const _Float16* aptr = Ah + (size_t)(rowBase + m) * lda + (hi ? 16 : 0);
  const _Float16* wrow = Wh + (size_t)(colBase + m) * Kpad + (hi ? 16 : 0);
  v8f acc = {};
  acc = wmmaAccum(aptr, wrow, Kpad, acc);
  const int col = colBase + m;
#pragma unroll
  for (int v = 0; v < 8; ++v) {
    const int row = rowBase + v + (hi ? 8 : 0);
    float o = 0.f;
    if (col < Nout) {
      o = acc[v] + (bias ? bias[col] : 0.f);
      if (act == ACT_LRELU)     o = lrelu_f(o);
      else if (act == ACT_RELU) o = o > 0.f ? o : 0.f;
    }
    Ch[(size_t)row * ldc + col] = (_Float16)o;
  }
}

// ---------------- GEMM: f16 A, f16 W, f32 out (GRU gates, head) -------------
__global__ void __launch_bounds__(128)
gemm_f16_f32out_kernel(const _Float16* __restrict__ Ah, int lda, int M, int Kpad,
                       const _Float16* __restrict__ Wh, const float* __restrict__ bias,
                       int Nout, int act, float* __restrict__ C, int ldc) {
  const int lane = threadIdx.x & 31;
  const int m = lane & 15;
  const bool hi = lane >= 16;
  const int rowBase = (blockIdx.x * 4 + (threadIdx.x >> 5)) * 16;
  const int colBase = blockIdx.y * 16;
  if (rowBase >= M) return;
  const _Float16* aptr = Ah + (size_t)(rowBase + m) * lda + (hi ? 16 : 0);
  const _Float16* wrow = Wh + (size_t)(colBase + m) * Kpad + (hi ? 16 : 0);
  v8f acc = {};
  acc = wmmaAccum(aptr, wrow, Kpad, acc);
  const int col = colBase + m;
#pragma unroll
  for (int v = 0; v < 8; ++v) {
    const int row = rowBase + v + (hi ? 8 : 0);
    if (col < ldc) {
      float o = 0.f;
      if (col < Nout) {
        o = acc[v] + (bias ? bias[col] : 0.f);
        if (act == ACT_LRELU)     o = lrelu_f(o);
        else if (act == ACT_RELU) o = o > 0.f ? o : 0.f;
      }
      C[(size_t)row * ldc + col] = o;
    }
  }
}

// ---- GATEConv GEMM1: A row e = concat(xh[src[e]], eattrh[e]) ---------------
// 6 col tiles per wave (cols 0..95 incl. zero padding of mjh).  grid = E/64.
__global__ void __launch_bounds__(128)
edge_gemm1_kernel(const _Float16* __restrict__ xh, const _Float16* __restrict__ eattrh,
                  const int* __restrict__ src, const _Float16* __restrict__ W1,
                  const _Float16* __restrict__ W2, _Float16* __restrict__ mjh, int E) {
  const int lane = threadIdx.x & 31;
  const int m = lane & 15;
  const bool hi = lane >= 16;
  const int rowBase = (blockIdx.x * 4 + (threadIdx.x >> 5)) * 16;
  if (rowBase >= E) return;
  const int e = rowBase + m;
  const int s = src[e];
  const _Float16* aptrX = xh     + (size_t)s * HPAD + (hi ? 16 : 0);
  const _Float16* aptrE = eattrh + (size_t)e * EPAD + (hi ? 16 : 0);
  const _Float16* wcol1 = W1 + (size_t)m * HPAD + (hi ? 16 : 0);
  const _Float16* wcol2 = W2 + (size_t)m * EPAD + (hi ? 16 : 0);
  v8f acc[6];
#pragma unroll
  for (int t = 0; t < 6; ++t) { v8f z = {}; acc[t] = z; }
  wmmaAccumMulti<6>(aptrX, wcol1, HPAD, acc);   // x[src] part, K 0..74
  wmmaAccumMulti<6>(aptrE, wcol2, EPAD, acc);   // edge_attr part, K 75..90
#pragma unroll
  for (int t = 0; t < 6; ++t) {
    const int col = t * 16 + m;
#pragma unroll
    for (int v = 0; v < 8; ++v) {
      const int row = rowBase + v + (hi ? 8 : 0);
      float o = (col < 75) ? lrelu_f(acc[t][v]) : 0.f;
      mjh[(size_t)row * HPAD + col] = (_Float16)o;
    }
  }
}

// ---- GATEConv GEMM2: (mj @ W^T)*softmax scattered into hacc[dst] -----------
// 5 col tiles per wave: mjh (HBM-resident, 153MB) read exactly once.
__global__ void __launch_bounds__(128)
edge_gemm2_scatter_kernel(const _Float16* __restrict__ mjh, const _Float16* __restrict__ W,
                          const float* __restrict__ eexp, const float* __restrict__ segs,
                          const int* __restrict__ dst, float* __restrict__ hacc, int E) {
  const int lane = threadIdx.x & 31;
  const int m = lane & 15;
  const bool hi = lane >= 16;
  const int rowBase = (blockIdx.x * 4 + (threadIdx.x >> 5)) * 16;
  if (rowBase >= E) return;
  const _Float16* aptr = mjh + (size_t)(rowBase + m) * HPAD + (hi ? 16 : 0);
  const _Float16* wcol = W + (size_t)m * HPAD + (hi ? 16 : 0);
  v8f acc[5];
#pragma unroll
  for (int t = 0; t < 5; ++t) { v8f z = {}; acc[t] = z; }
  wmmaAccumMulti<5>(aptr, wcol, HPAD, acc);
#pragma unroll
  for (int t = 0; t < 5; ++t) {
    const int col = t * 16 + m;
    if (col < 75) {
#pragma unroll
      for (int v = 0; v < 8; ++v) {
        const int e = rowBase + v + (hi ? 8 : 0);
        const int d = dst[e];
        atomicAdd(&hacc[(size_t)d * HPAD + col], acc[t][v] * (eexp[e] / segs[d]));
      }
    }
  }
}

// ----------------------------- packing -------------------------------------
__global__ void wpack_kernel(const float* __restrict__ W, int ldw, int Koff,
                             int Kact, int Nout, _Float16* __restrict__ Wh,
                             int Kpad, int NoutPad) {
  int i = blockIdx.x * blockDim.x + threadIdx.x;
  if (i >= NoutPad * Kpad) return;
  int col = i / Kpad, k = i % Kpad;
  float v = (col < Nout && k < Kact) ? W[(size_t)col * ldw + Koff + k] : 0.f;
  Wh[i] = (_Float16)v;
}

__global__ void xpack_kernel(const float* __restrict__ X, _Float16* __restrict__ Xh, int n) {
  int i = blockIdx.x * blockDim.x + threadIdx.x;
  if (i < n) Xh[i] = (_Float16)X[i];
}

__global__ void eattr_pack_kernel(const float* __restrict__ ea, _Float16* __restrict__ eh, int E) {
  int i = blockIdx.x * blockDim.x + threadIdx.x;
  if (i >= E * EPAD) return;
  int e = i / EPAD, c = i % EPAD;
  eh[i] = (_Float16)((c < 16) ? ea[(size_t)e * 16 + c] : 0.f);
}

// ----------------------------- elementwise ---------------------------------
__global__ void fill_kernel(float* __restrict__ p, float v, int n) {
  int i = blockIdx.x * blockDim.x + threadIdx.x;
  if (i < n) p[i] = v;
}

__global__ void init_bias_rows_kernel(float* __restrict__ h, const float* __restrict__ bias, int M) {
  int i = blockIdx.x * blockDim.x + threadIdx.x;
  if (i < M * HPAD) { int c = i % HPAD; h[i] = (c < 75) ? bias[c] : 0.f; }
}

__global__ void elu_pack_kernel(const float* __restrict__ hacc, _Float16* __restrict__ hh, int M) {
  int i = blockIdx.x * blockDim.x + threadIdx.x;
  if (i >= M * HPAD) return;
  int c = i % HPAD;
  float o = 0.f;
  if (c < 75) { float v = hacc[i]; o = v > 0.f ? v : __expf(v) - 1.f; }
  hh[i] = (_Float16)o;
}

__global__ void relu_pack_kernel(const float* __restrict__ a, _Float16* __restrict__ oh, int n) {
  int i = blockIdx.x * blockDim.x + threadIdx.x;
  if (i < n) { float v = a[i]; oh[i] = (_Float16)(v > 0.f ? v : 0.f); }
}

__global__ void node_dot_kernel(const _Float16* __restrict__ X, int ld,
                                const float* __restrict__ v, float* __restrict__ s, int M) {
  int n = blockIdx.x * blockDim.x + threadIdx.x;
  if (n >= M) return;
  float acc = 0.f;
#pragma unroll
  for (int c = 0; c < 75; ++c) acc += (float)X[(size_t)n * ld + c] * v[c];
  s[n] = acc;
}

__global__ void edge_logit_gate_kernel(const _Float16* __restrict__ mjh,
                                       const float* __restrict__ attl,
                                       const float* __restrict__ sB,
                                       const int* __restrict__ dst,
                                       float* __restrict__ elog, int E) {
  int e = blockIdx.x * blockDim.x + threadIdx.x;
  if (e >= E) return;
  float acc = 0.f;
#pragma unroll
  for (int c = 0; c < 75; ++c) acc += (float)mjh[(size_t)e * HPAD + c] * attl[c];
  elog[e] = lrelu_f(acc + sB[dst[e]]);
}

__global__ void edge_logit_atom_kernel(const float* __restrict__ sA, const float* __restrict__ sB,
                                       const int* __restrict__ src, const int* __restrict__ dst,
                                       float* __restrict__ elog, int E) {
  int e = blockIdx.x * blockDim.x + threadIdx.x;
  if (e >= E) return;
  elog[e] = lrelu_f(sA[src[e]] + sB[dst[e]]);
}

__global__ void node_logit_mol_kernel(const float* __restrict__ sA, const float* __restrict__ dg,
                                      const int* __restrict__ batch, float* __restrict__ elog, int N) {
  int n = blockIdx.x * blockDim.x + threadIdx.x;
  if (n >= N) return;
  elog[n] = lrelu_f(sA[n] + dg[batch[n]]);
}

__global__ void seg_max_kernel(const float* __restrict__ logit, const int* __restrict__ seg,
                               float* __restrict__ m, int n) {
  int i = blockIdx.x * blockDim.x + threadIdx.x;
  if (i < n) atomicMaxFloat(&m[seg[i]], logit[i]);
}

__global__ void seg_expsum_kernel(const float* __restrict__ logit, const int* __restrict__ seg,
                                  const float* __restrict__ m, float* __restrict__ e,
                                  float* __restrict__ s, int n) {
  int i = blockIdx.x * blockDim.x + threadIdx.x;
  if (i >= n) return;
  int g = seg[i];
  float ev = __expf(logit[i] - m[g]);
  e[i] = ev;
  atomicAdd(&s[g], ev);
}

__global__ void edge_scatter_weighted_kernel(const _Float16* __restrict__ xlh,
                                             const int* __restrict__ src, const int* __restrict__ dst,
                                             const float* __restrict__ eexp, const float* __restrict__ segs,
                                             float* __restrict__ hacc, int E) {
  int i = blockIdx.x * blockDim.x + threadIdx.x;
  if (i >= E * HPAD) return;
  int e = i / HPAD, c = i % HPAD;
  if (c >= 75) return;
  int d = dst[e];
  atomicAdd(&hacc[(size_t)d * HPAD + c],
            (float)xlh[(size_t)src[e] * HPAD + c] * (eexp[e] / segs[d]));
}

__global__ void node_scatter_sum_kernel(const _Float16* __restrict__ xh, const int* __restrict__ batch,
                                        float* __restrict__ out, int N) {
  int i = blockIdx.x * blockDim.x + threadIdx.x;
  if (i >= N * HPAD) return;
  int n = i / HPAD, c = i % HPAD;
  if (c >= 75) return;
  atomicAdd(&out[(size_t)batch[n] * HPAD + c], (float)xh[i]);
}

__global__ void node_scatter_weighted_kernel(const _Float16* __restrict__ xlh, const int* __restrict__ batch,
                                             const float* __restrict__ eexp, const float* __restrict__ sg,
                                             float* __restrict__ hg, int N) {
  int i = blockIdx.x * blockDim.x + threadIdx.x;
  if (i >= N * HPAD) return;
  int n = i / HPAD, c = i % HPAD;
  if (c >= 75) return;
  int g = batch[n];
  atomicAdd(&hg[(size_t)g * HPAD + c], (float)xlh[i] * (eexp[n] / sg[g]));
}

__global__ void gru_combine_kernel(const float* __restrict__ gi, const float* __restrict__ gh,
                                   const _Float16* __restrict__ xold, _Float16* __restrict__ xnew, int M) {
  int i = blockIdx.x * blockDim.x + threadIdx.x;
  if (i >= M * HPAD) return;
  int row = i / HPAD, c = i % HPAD;
  if (c >= 75) { xnew[i] = (_Float16)0.f; return; }
  size_t b = (size_t)row * GPAD;
  float r  = sigmoid_f(gi[b + c]      + gh[b + c]);
  float z  = sigmoid_f(gi[b + 75 + c] + gh[b + 75 + c]);
  float nn = tanhf(gi[b + 150 + c] + r * gh[b + 150 + c]);
  float v  = (1.f - z) * nn + z * (float)xold[i];
  xnew[i]  = (_Float16)(v > 0.f ? v : 0.f);
}

// ---------------------------------------------------------------------------
extern "C" void kernel_launch(void* const* d_in, const int* in_sizes, int n_in,
                              void* d_out, int out_size, void* d_ws, size_t ws_size,
                              hipStream_t stream) {
  (void)n_in; (void)ws_size;
  const float* x           = (const float*)d_in[0];
  const int*   eidx        = (const int*)  d_in[1];
  const float* eattr       = (const float*)d_in[2];
  const int*   batch       = (const int*)  d_in[3];
  const float* lin1_W      = (const float*)d_in[4];
  const float* lin1_b      = (const float*)d_in[5];
  const float* gate_att_l  = (const float*)d_in[6];
  const float* gate_att_r  = (const float*)d_in[7];
  const float* gate_lin1_W = (const float*)d_in[8];
  const float* gate_lin2_W = (const float*)d_in[9];
  const float* gate_bias   = (const float*)d_in[10];
  const float* gru0_Wih    = (const float*)d_in[11];
  const float* gru0_Whh    = (const float*)d_in[12];
  const float* gru0_bih    = (const float*)d_in[13];
  const float* gru0_bhh    = (const float*)d_in[14];
  const float* atom_lin_W  = (const float*)d_in[15];
  const float* atom_att_src= (const float*)d_in[16];
  const float* atom_att_dst= (const float*)d_in[17];
  const float* atom_bias   = (const float*)d_in[18];
  const float* atom_gru_Wih= (const float*)d_in[19];
  const float* atom_gru_Whh= (const float*)d_in[20];
  const float* atom_gru_bih= (const float*)d_in[21];
  const float* atom_gru_bhh= (const float*)d_in[22];
  const float* mol_lin_W   = (const float*)d_in[23];
  const float* mol_att_src = (const float*)d_in[24];
  const float* mol_att_dst = (const float*)d_in[25];
  const float* mol_bias    = (const float*)d_in[26];
  const float* mol_gru_Wih = (const float*)d_in[27];
  const float* mol_gru_Whh = (const float*)d_in[28];
  const float* mol_gru_bih = (const float*)d_in[29];
  const float* mol_gru_bhh = (const float*)d_in[30];
  const float* lin2_W      = (const float*)d_in[31];
  const float* lin2_b      = (const float*)d_in[32];
  const float* mlp_W1      = (const float*)d_in[33];
  const float* mlp_b1      = (const float*)d_in[34];
  const float* mlp_W2      = (const float*)d_in[35];
  const float* mlp_b2      = (const float*)d_in[36];

  const int N = in_sizes[0] / 64;   // 50000
  const int E = in_sizes[1] / 2;    // 800000
  const int G = out_size / 50;      // 1024
  const int* src = eidx;
  const int* dst = eidx + E;

  // ---- workspace carve ----
  char* ws = (char*)d_ws;
  size_t off = 0;
  auto alloc = [&](size_t bytes) -> void* {
    void* p = ws + off; off += (bytes + 255) & ~(size_t)255; return p;
  };
  _Float16* xh0    = (_Float16*)alloc((size_t)N * 64 * 2);
  _Float16* xh     = (_Float16*)alloc((size_t)N * HPAD * 2);
  _Float16* hh     = (_Float16*)alloc((size_t)N * HPAD * 2);
  _Float16* xlh    = (_Float16*)alloc((size_t)N * HPAD * 2);
  float*    hacc   = (float*)alloc((size_t)N * HPAD * 4);
  _Float16* mjh    = (_Float16*)alloc((size_t)E * HPAD * 2);   // gi/gh alias later
  _Float16* eattrh = (_Float16*)alloc((size_t)E * EPAD * 2);
  float*    elog   = (float*)alloc((size_t)E * 4);
  float*    eexp   = (float*)alloc((size_t)E * 4);
  float*    segm   = (float*)alloc((size_t)N * 4);
  float*    segs   = (float*)alloc((size_t)N * 4);
  float*    sA     = (float*)alloc((size_t)N * 4);
  float*    sB     = (float*)alloc((size_t)N * 4);
  float*    ogacc  = (float*)alloc((size_t)G * HPAD * 4);
  _Float16* outh   = (_Float16*)alloc((size_t)G * HPAD * 2);
  _Float16* olh    = (_Float16*)alloc((size_t)G * HPAD * 2);
  float*    hgacc  = (float*)alloc((size_t)G * HPAD * 4);
  _Float16* hgh    = (_Float16*)alloc((size_t)G * HPAD * 2);
  float*    gig    = (float*)alloc((size_t)G * GPAD * 4);
  float*    ghg    = (float*)alloc((size_t)G * GPAD * 4);
  float*    mg     = (float*)alloc((size_t)G * 4);
  float*    sg     = (float*)alloc((size_t)G * 4);
  float*    dgv    = (float*)alloc((size_t)G * 4);
  _Float16* t1h    = (_Float16*)alloc((size_t)G * HPAD * 2);
  _Float16* t2h    = (_Float16*)alloc((size_t)G * 64 * 2);
  // f16 packed weights: all H-out weights packed with 96 rows so column tiles
  // up to col 95 (ldc=96 launches / 6-tile edge kernel) stay in-bounds.
  _Float16* wh_lin1 = (_Float16*)alloc(96 * 64 * 2);
  _Float16* wh_g1a  = (_Float16*)alloc(96 * HPAD * 2);
  _Float16* wh_g1b  = (_Float16*)alloc(96 * EPAD * 2);
  _Float16* wh_g2   = (_Float16*)alloc(96 * HPAD * 2);
  _Float16* wh_g0i  = (_Float16*)alloc(GPAD * HPAD * 2);
  _Float16* wh_g0h  = (_Float16*)alloc(GPAD * HPAD * 2);
  _Float16* wh_al[3], *wh_ai[3], *wh_ah[3];
  for (int l = 0; l < 3; ++l) {
    wh_al[l] = (_Float16*)alloc(96 * HPAD * 2);
    wh_ai[l] = (_Float16*)alloc(GPAD * HPAD * 2);
    wh_ah[l] = (_Float16*)alloc(GPAD * HPAD * 2);
  }
  _Float16* wh_mol  = (_Float16*)alloc(96 * HPAD * 2);
  _Float16* wh_mi   = (_Float16*)alloc(GPAD * HPAD * 2);
  _Float16* wh_mh   = (_Float16*)alloc(GPAD * HPAD * 2);
  _Float16* wh_lin2 = (_Float16*)alloc(96 * HPAD * 2);
  _Float16* wh_mlp1 = (_Float16*)alloc(64 * HPAD * 2);
  _Float16* wh_mlp2 = (_Float16*)alloc(64 * 64 * 2);
  // GRU gate buffers alias the (then-dead) mjh region: 2*N*240*4 = 96MB < 153.6MB
  float* gi = (float*)mjh;
  float* gh = gi + (size_t)N * GPAD;

  auto gemmGrid = [](int M, int ldc) { return dim3((unsigned)((M + 63) / 64), (unsigned)(ldc / 16)); };
  auto edgeGrid = [](int M) { return dim3((unsigned)((M + 63) / 64)); };
  auto lin = [](long long n) { return dim3((unsigned)((n + 255) / 256)); };
  const dim3 TB(256);

  // ---- pack weights + inputs to f16 ----
  auto wp = [&](const float* W, int ldw, int Koff, int Kact, int Nout,
                _Float16* Wh, int Kpad, int NoutPad) {
    wpack_kernel<<<lin((long long)NoutPad * Kpad), TB, 0, stream>>>(W, ldw, Koff, Kact, Nout, Wh, Kpad, NoutPad);
  };
  wp(lin1_W, 64, 0, 64, 75, wh_lin1, 64, 96);
  wp(gate_lin1_W, 91, 0, 75, 75, wh_g1a, HPAD, 96);
  wp(gate_lin1_W, 91, 75, 16, 75, wh_g1b, EPAD, 96);
  wp(gate_lin2_W, 75, 0, 75, 75, wh_g2, HPAD, 96);
  wp(gru0_Wih, 75, 0, 75, 225, wh_g0i, HPAD, GPAD);
  wp(gru0_Whh, 75, 0, 75, 225, wh_g0h, HPAD, GPAD);
  for (int l = 0; l < 3; ++l) {
    wp(atom_lin_W  + (size_t)l * 75 * 75,  75, 0, 75, 75,  wh_al[l], HPAD, 96);
    wp(atom_gru_Wih+ (size_t)l * 225 * 75, 75, 0, 75, 225, wh_ai[l], HPAD, GPAD);
    wp(atom_gru_Whh+ (size_t)l * 225 * 75, 75, 0, 75, 225, wh_ah[l], HPAD, GPAD);
  }
  wp(mol_lin_W, 75, 0, 75, 75, wh_mol, HPAD, 96);
  wp(mol_gru_Wih, 75, 0, 75, 225, wh_mi, HPAD, GPAD);
  wp(mol_gru_Whh, 75, 0, 75, 225, wh_mh, HPAD, GPAD);
  wp(lin2_W, 75, 0, 75, 75, wh_lin2, HPAD, 96);
  wp(mlp_W1, 75, 0, 75, 50, wh_mlp1, HPAD, 64);
  wp(mlp_W2, 50, 0, 50, 50, wh_mlp2, 64, 64);
  xpack_kernel<<<lin((long long)N * 64), TB, 0, stream>>>(x, xh0, N * 64);
  eattr_pack_kernel<<<lin((long long)E * EPAD), TB, 0, stream>>>(eattr, eattrh, E);

  // ---- lin1 ----
  gemm_f16_kernel<<<gemmGrid(N, HPAD), 128, 0, stream>>>(xh0, 64, N, 64, wh_lin1, lin1_b, 75, ACT_LRELU, xh, HPAD);

  // ---- GATEConv ----
  edge_gemm1_kernel<<<edgeGrid(E), 128, 0, stream>>>(xh, eattrh, src, wh_g1a, wh_g1b, mjh, E);
  node_dot_kernel<<<lin(N), TB, 0, stream>>>(xh, HPAD, gate_att_r, sB, N);
  edge_logit_gate_kernel<<<lin(E), TB, 0, stream>>>(mjh, gate_att_l, sB, dst, elog, E);
  fill_kernel<<<lin(N), TB, 0, stream>>>(segm, -1e30f, N);
  fill_kernel<<<lin(N), TB, 0, stream>>>(segs, 0.f, N);
  seg_max_kernel<<<lin(E), TB, 0, stream>>>(elog, dst, segm, E);
  seg_expsum_kernel<<<lin(E), TB, 0, stream>>>(elog, dst, segm, eexp, segs, E);
  init_bias_rows_kernel<<<lin((long long)N * HPAD), TB, 0, stream>>>(hacc, gate_bias, N);
  edge_gemm2_scatter_kernel<<<edgeGrid(E), 128, 0, stream>>>(mjh, wh_g2, eexp, segs, dst, hacc, E);
  elu_pack_kernel<<<lin((long long)N * HPAD), TB, 0, stream>>>(hacc, hh, N);
  gemm_f16_f32out_kernel<<<gemmGrid(N, GPAD), 128, 0, stream>>>(hh, HPAD, N, HPAD, wh_g0i, gru0_bih, 225, ACT_NONE, gi, GPAD);
  gemm_f16_f32out_kernel<<<gemmGrid(N, GPAD), 128, 0, stream>>>(xh, HPAD, N, HPAD, wh_g0h, gru0_bhh, 225, ACT_NONE, gh, GPAD);
  gru_combine_kernel<<<lin((long long)N * HPAD), TB, 0, stream>>>(gi, gh, xh, xh, N);

  // ---- atom GAT + GRU layers ----
  for (int l = 0; l < 3; ++l) {
    const float* asrc = atom_att_src + (size_t)l * 75;
    const float* adst = atom_att_dst + (size_t)l * 75;
    const float* ab   = atom_bias    + (size_t)l * 75;
    const float* bih  = atom_gru_bih + (size_t)l * 225;
    const float* bhh  = atom_gru_bhh + (size_t)l * 225;
    gemm_f16_kernel<<<gemmGrid(N, HPAD), 128, 0, stream>>>(xh, HPAD, N, HPAD, wh_al[l], nullptr, 75, ACT_NONE, xlh, HPAD);
    node_dot_kernel<<<lin(N), TB, 0, stream>>>(xlh, HPAD, asrc, sA, N);
    node_dot_kernel<<<lin(N), TB, 0, stream>>>(xlh, HPAD, adst, sB, N);
    edge_logit_atom_kernel<<<lin(E), TB, 0, stream>>>(sA, sB, src, dst, elog, E);
    fill_kernel<<<lin(N), TB, 0, stream>>>(segm, -1e30f, N);
    fill_kernel<<<lin(N), TB, 0, stream>>>(segs, 0.f, N);
    seg_max_kernel<<<lin(E), TB, 0, stream>>>(elog, dst, segm, E);
    seg_expsum_kernel<<<lin(E), TB, 0, stream>>>(elog, dst, segm, eexp, segs, E);
    init_bias_rows_kernel<<<lin((long long)N * HPAD), TB, 0, stream>>>(hacc, ab, N);
    edge_scatter_weighted_kernel<<<lin((long long)E * HPAD), TB, 0, stream>>>(xlh, src, dst, eexp, segs, hacc, E);
    elu_pack_kernel<<<lin((long long)N * HPAD), TB, 0, stream>>>(hacc, hh, N);
    gemm_f16_f32out_kernel<<<gemmGrid(N, GPAD), 128, 0, stream>>>(hh, HPAD, N, HPAD, wh_ai[l], bih, 225, ACT_NONE, gi, GPAD);
    gemm_f16_f32out_kernel<<<gemmGrid(N, GPAD), 128, 0, stream>>>(xh, HPAD, N, HPAD, wh_ah[l], bhh, 225, ACT_NONE, gh, GPAD);
    gru_combine_kernel<<<lin((long long)N * HPAD), TB, 0, stream>>>(gi, gh, xh, xh, N);
  }

  // ---- molecule embedding ----
  fill_kernel<<<lin((long long)G * HPAD), TB, 0, stream>>>(ogacc, 0.f, G * HPAD);
  node_scatter_sum_kernel<<<lin((long long)N * HPAD), TB, 0, stream>>>(xh, batch, ogacc, N);
  relu_pack_kernel<<<lin((long long)G * HPAD), TB, 0, stream>>>(ogacc, outh, G * HPAD);
  gemm_f16_kernel<<<gemmGrid(N, HPAD), 128, 0, stream>>>(xh, HPAD, N, HPAD, wh_mol, nullptr, 75, ACT_NONE, xlh, HPAD);
  node_dot_kernel<<<lin(N), TB, 0, stream>>>(xlh, HPAD, mol_att_src, sA, N);
  for (int t = 0; t < 4; ++t) {
    gemm_f16_kernel<<<gemmGrid(G, HPAD), 128, 0, stream>>>(outh, HPAD, G, HPAD, wh_mol, nullptr, 75, ACT_NONE, olh, HPAD);
    node_dot_kernel<<<lin(G), TB, 0, stream>>>(olh, HPAD, mol_att_dst, dgv, G);
    node_logit_mol_kernel<<<lin(N), TB, 0, stream>>>(sA, dgv, batch, elog, N);
    fill_kernel<<<lin(G), TB, 0, stream>>>(mg, -1e30f, G);
    fill_kernel<<<lin(G), TB, 0, stream>>>(sg, 0.f, G);
    seg_max_kernel<<<lin(N), TB, 0, stream>>>(elog, batch, mg, N);
    seg_expsum_kernel<<<lin(N), TB, 0, stream>>>(elog, batch, mg, eexp, sg, N);
    init_bias_rows_kernel<<<lin((long long)G * HPAD), TB, 0, stream>>>(hgacc, mol_bias, G);
    node_scatter_weighted_kernel<<<lin((long long)N * HPAD), TB, 0, stream>>>(xlh, batch, eexp, sg, hgacc, N);
    elu_pack_kernel<<<lin((long long)G * HPAD), TB, 0, stream>>>(hgacc, hgh, G);
    gemm_f16_f32out_kernel<<<gemmGrid(G, GPAD), 128, 0, stream>>>(hgh, HPAD, G, HPAD, wh_mi, mol_gru_bih, 225, ACT_NONE, gig, GPAD);
    gemm_f16_f32out_kernel<<<gemmGrid(G, GPAD), 128, 0, stream>>>(outh, HPAD, G, HPAD, wh_mh, mol_gru_bhh, 225, ACT_NONE, ghg, GPAD);
    gru_combine_kernel<<<lin((long long)G * HPAD), TB, 0, stream>>>(gig, ghg, outh, outh, G);
  }

  // ---- head ----
  gemm_f16_kernel<<<gemmGrid(G, HPAD), 128, 0, stream>>>(outh, HPAD, G, HPAD, wh_lin2, lin2_b, 75, ACT_NONE, t1h, HPAD);
  gemm_f16_kernel<<<gemmGrid(G, 64), 128, 0, stream>>>(t1h, HPAD, G, HPAD, wh_mlp1, mlp_b1, 50, ACT_RELU, t2h, 64);
  gemm_f16_f32out_kernel<<<gemmGrid(G, 64), 128, 0, stream>>>(t2h, 64, G, 64, wh_mlp2, mlp_b2, 50, ACT_RELU, (float*)d_out, 50);
}
